// TestModule_80066780332332
// MI455X (gfx1250) — matrix operand order repaired and checked by reading, not compiled, not run on hardware
//
#include <hip/hip_runtime.h>

// Problem constants (match reference)
#define VOCAB 90000
#define DIM   768
#define CTX   1024
#define ACD   10
#define KSYN  4
#define DTOT  778          // DIM + ACD
#define KPAD  800          // DTOT padded to multiple of 32
#define NK    25           // KPAD / 32
#define BM    128
#define BN    128
#define NPAD  90112        // 704 * 128
#define LDST  40           // LDS row stride in halfs (80B, 16B aligned, low-conflict)

typedef _Float16 v8h  __attribute__((ext_vector_type(8)));
typedef _Float16 v16h __attribute__((ext_vector_type(16)));
typedef float    v8f  __attribute__((ext_vector_type(8)));
typedef int      v4i  __attribute__((vector_size(16)));   // matches builtin param type

union V16H { struct { v8h lo, hi; } p; v16h v; };

#if defined(__gfx1250__) && __has_builtin(__builtin_amdgcn_global_load_async_to_lds_b128)
#define HAVE_ASYNC_LDS 1
#else
#define HAVE_ASYNC_LDS 0
#endif

__device__ __forceinline__ void wait_async0() {
#if __has_builtin(__builtin_amdgcn_s_wait_asynccnt)
  __builtin_amdgcn_s_wait_asynccnt(0);
#else
  asm volatile("s_wait_asynccnt 0x0" ::: "memory");
#endif
}

// 16-byte global -> LDS copy (async on CDNA5, sync fallback otherwise)
__device__ __forceinline__ void cp16(const void* g, void* l) {
#if HAVE_ASYNC_LDS
  __builtin_amdgcn_global_load_async_to_lds_b128(
      (__attribute__((address_space(1))) v4i*)g,
      (__attribute__((address_space(3))) v4i*)l, 0, 0);
#else
  *(uint4*)l = *(const uint4*)g;
#endif
}

// ---------------------------------------------------------------------------
// Kernel 1: build A = f16(max(base_emb, masked synonym embs) ++ padding), K-padded
// ---------------------------------------------------------------------------
__global__ __launch_bounds__(256) void build_a_kernel(
    const int* __restrict__ ids, const float* __restrict__ emb,
    const int* __restrict__ syn_ids, const unsigned char* __restrict__ syn_mask,
    const float* __restrict__ padding, _Float16* __restrict__ Ah)
{
  const int t   = blockIdx.x;      // token 0..1023
  const int tid = threadIdx.x;
  const int sid = ids[t];

  int s[KSYN]; int mk[KSYN];
#pragma unroll
  for (int j = 0; j < KSYN; ++j) {
    s[j]  = syn_ids[sid * KSYN + j];
    mk[j] = syn_mask[sid * KSYN + j];
  }
  const float* base = emb + (size_t)sid * DIM;

  for (int d = tid; d < KPAD; d += 256) {
    _Float16 o;
    if (d < DIM) {
      float v = base[d];
#pragma unroll
      for (int j = 0; j < KSYN; ++j) {
        float sv = emb[(size_t)s[j] * DIM + d];
        if (mk[j]) v = fmaxf(v, sv);
      }
      o = (_Float16)v;
    } else if (d < DTOT) {
      o = (_Float16)padding[t * ACD + (d - DIM)];
    } else {
      o = (_Float16)0.0f;               // K padding -> zero so B pad is don't-care
    }
    Ah[(size_t)t * KPAD + d] = o;
  }
}

// ---------------------------------------------------------------------------
// Kernel 2: Bt[n][k] = f16(rev[k][n]), n-major (transposed), zero padded.
// ---------------------------------------------------------------------------
__global__ __launch_bounds__(256) void convert_b_kernel(
    const float* __restrict__ rev, _Float16* __restrict__ Bt)
{
  const int k0 = blockIdx.x * 32;       // 25 blocks
  const int n0 = blockIdx.y * 32;       // 2816 blocks
  __shared__ _Float16 tile[32][33];
  const int t = threadIdx.x;
#pragma unroll
  for (int i = 0; i < 4; ++i) {
    int idx = t + i * 256;
    int kl = idx >> 5, nl = idx & 31;
    int k = k0 + kl, n = n0 + nl;
    float v = (k < DTOT && n < VOCAB) ? rev[(size_t)k * VOCAB + n] : 0.0f;
    tile[nl][kl] = (_Float16)v;
  }
  __syncthreads();
#pragma unroll
  for (int i = 0; i < 4; ++i) {
    int idx = t + i * 256;
    int nl = idx >> 5, kl = idx & 31;
    Bt[(size_t)(n0 + nl) * KPAD + k0 + kl] = tile[nl][kl];
  }
}

// ---------------------------------------------------------------------------
// Kernel 3: C[1024,90000] = A x B, f16 WMMA / f32 accum, async-LDS double buffer.
// 128x128 tile / block, 8 waves, wave tile 32x64 (2x4 WMMA 16x16 tiles).
// ---------------------------------------------------------------------------
template <bool USE_BT>
__global__ __launch_bounds__(256) void gemm_kernel(
    const _Float16* __restrict__ Ah, const _Float16* __restrict__ Bt,
    const float* __restrict__ rev, float* __restrict__ out)
{
  __shared__ _Float16 Als[2][BM][LDST];
  __shared__ _Float16 Bls[2][BN][LDST];

  const int tid  = threadIdx.x;
  const int lane = tid & 31;
  const int wave = tid >> 5;
  const int wm   = wave & 3;            // wave row  (4)
  const int wn   = wave >> 2;           // wave col  (2)
  const int m0   = blockIdx.x * BM;     // m fastest -> consecutive blocks share B in L2
  const int n0   = blockIdx.y * BN;
  const int lm   = lane & 15;
  const int lh   = lane >> 4;

  // Tile loader addressing: 512 x 16B chunks per tile, 2 per thread per matrix.
  const int r0 = tid >> 2;              // 0..63
  const int c8 = (tid & 3) * 8;         // half offset 0/8/16/24

  v8f acc[2][4];
#pragma unroll
  for (int i = 0; i < 2; ++i)
#pragma unroll
    for (int j = 0; j < 4; ++j)
      acc[i][j] = (v8f){0.f, 0.f, 0.f, 0.f, 0.f, 0.f, 0.f, 0.f};

  auto load_tile = [&](int buf, int kk) {
    const int k0 = kk * 32;
    cp16(Ah + (size_t)(m0 + r0) * KPAD + k0 + c8,      &Als[buf][r0][c8]);
    cp16(Ah + (size_t)(m0 + r0 + 64) * KPAD + k0 + c8, &Als[buf][r0 + 64][c8]);
    if constexpr (USE_BT) {
      cp16(Bt + (size_t)(n0 + r0) * KPAD + k0 + c8,      &Bls[buf][r0][c8]);
      cp16(Bt + (size_t)(n0 + r0 + 64) * KPAD + k0 + c8, &Bls[buf][r0 + 64][c8]);
    } else {
      // inline fp32 -> f16 transpose path (only if workspace too small for Bt)
      const int fn  = tid & 127;
      const int fkb = (tid >> 7) * 16;
      const int n   = n0 + fn;
#pragma unroll
      for (int i = 0; i < 16; ++i) {
        int k = k0 + fkb + i;
        float v = (k < DTOT && n < VOCAB) ? rev[(size_t)k * VOCAB + n] : 0.0f;
        Bls[buf][fn][fkb + i] = (_Float16)v;   // immediate store, no liveness
      }
    }
  };

  auto compute = [&](int cur) {
    V16H af[2], bf[4];
#pragma unroll
    for (int mt = 0; mt < 2; ++mt) {
      const _Float16* ap = &Als[cur][wm * 32 + mt * 16 + lm][0];
      // A 16x32 f16 frag: elems 0..7 -> K = lh*8 + j ; elems 8..15 -> K = 16 + lh*8 + j
      af[mt].p.lo = *(const v8h*)(ap + lh * 8);
      af[mt].p.hi = *(const v8h*)(ap + 16 + lh * 8);
    }
#pragma unroll
    for (int nt = 0; nt < 4; ++nt) {
      const _Float16* bp = &Bls[cur][wn * 64 + nt * 16 + lm][0];
      // B 32x16 f16 frag: elems 0..7 -> K = lh*8 + j ; elems 8..15 -> K = 16 + lh*8 + j  (lane holds column n)
      bf[nt].p.lo = *(const v8h*)(bp + lh * 8);
      bf[nt].p.hi = *(const v8h*)(bp + 16 + lh * 8);
    }
#pragma unroll
    for (int mt = 0; mt < 2; ++mt)
#pragma unroll
      for (int nt = 0; nt < 4; ++nt)
        acc[mt][nt] = __builtin_amdgcn_wmma_f32_16x16x32_f16(
            false, af[mt].v, false, bf[nt].v, (short)0, acc[mt][nt], false, false);
  };

  // ---- Pipeline: one barrier per k-step, async copies overlap compute ----
  load_tile(0, 0);
#if HAVE_ASYNC_LDS
  wait_async0();
#endif
  __syncthreads();

  for (int kk = 0; kk < NK; ++kk) {
    const int  cur  = kk & 1;
    const bool more = (kk + 1 < NK);
#if HAVE_ASYNC_LDS
    if (more) load_tile(cur ^ 1, kk + 1);   // async into other buffer
    compute(cur);
    if (more) wait_async0();                // my async LDS writes done
#else
    compute(cur);
    if (more) load_tile(cur ^ 1, kk + 1);   // sync copy, no cross-compute liveness
#endif
    __syncthreads();
  }

  // Store C: VGPR r -> M = lh*8 + r ; lane -> N = lm
#pragma unroll
  for (int mt = 0; mt < 2; ++mt)
#pragma unroll
    for (int nt = 0; nt < 4; ++nt) {
      const int row0 = m0 + wm * 32 + mt * 16 + lh * 8;
      const int col  = n0 + wn * 64 + nt * 16 + lm;
      if (col < VOCAB) {
#pragma unroll
        for (int r = 0; r < 8; ++r)
          out[(size_t)(row0 + r) * VOCAB + col] = acc[mt][nt][r];
      }
    }
}

// ---------------------------------------------------------------------------
extern "C" void kernel_launch(void* const* d_in, const int* in_sizes, int n_in,
                              void* d_out, int out_size, void* d_ws, size_t ws_size,
                              hipStream_t stream)
{
  (void)in_sizes; (void)n_in; (void)out_size;
  const int*           ids      = (const int*)d_in[0];
  const float*         emb      = (const float*)d_in[1];
  const float*         rev      = (const float*)d_in[2];
  const int*           syn_ids  = (const int*)d_in[3];
  const unsigned char* syn_mask = (const unsigned char*)d_in[4];
  const float*         padding  = (const float*)d_in[5];
  float*               out      = (float*)d_out;

  _Float16* Ah = (_Float16*)d_ws;
  const size_t ahBytes = (size_t)CTX * KPAD * sizeof(_Float16);      // 1.6 MB
  const size_t btBytes = (size_t)NPAD * KPAD * sizeof(_Float16);     // ~144 MB (L2-resident)
  _Float16* Bt = (_Float16*)((char*)d_ws + ahBytes);
  const bool useBt = (ws_size >= ahBytes + btBytes);

  build_a_kernel<<<CTX, 256, 0, stream>>>(ids, emb, syn_ids, syn_mask, padding, Ah);

  if (useBt) {
    convert_b_kernel<<<dim3(NK, NPAD / 32), 256, 0, stream>>>(rev, Bt);
    gemm_kernel<true><<<dim3(CTX / BM, NPAD / BN), 256, 0, stream>>>(Ah, Bt, rev, out);
  } else {
    gemm_kernel<false><<<dim3(CTX / BM, NPAD / BN), 256, 0, stream>>>(Ah, nullptr, rev, out);
  }
}
